// DynamicSpatialFcModule_68238440399456
// MI455X (gfx1250) — matrix operand order repaired and checked by reading, not compiled, not run on hardware
//
#include <hip/hip_runtime.h>
#include <hip/hip_bf16.h>

typedef __attribute__((ext_vector_type(16))) _Float16 v16h;
typedef __attribute__((ext_vector_type(8)))  _Float16 v8h;
typedef __attribute__((ext_vector_type(4)))  _Float16 v4h;
typedef __attribute__((ext_vector_type(8)))  float    v8f;
typedef __attribute__((ext_vector_type(4)))  float    v4f;

#define BATCH      64
#define PLANES     256
#define SPATIAL_IN 3136      // 56*56
#define S4         784       // SPATIAL_IN/4
#define HIDDEN     196
#define KEEP       1568      // SPATIAL_IN - REMOVED_NUM
#define N1PAD      208       // HIDDEN padded to multiple of 16 (GEMM1 N)
#define KPAD       224       // HIDDEN padded to multiple of 32 (GEMM2 K)

// ---------------------------------------------------------------------------
// 1) Channel-mean pool: x[64,256,3136] f32 -> pooledh[64,3136] f16
//    Pure streaming read (205 MB) -> nontemporal vector loads, coalesced.
// ---------------------------------------------------------------------------
__global__ void pool_kernel(const float* __restrict__ x,
                            _Float16* __restrict__ pooledh) {
    int t  = blockIdx.x * blockDim.x + threadIdx.x;   // 0 .. 64*784-1
    int b  = t / S4;
    int s4 = t % S4;
    const v4f* xb = (const v4f*)x + (size_t)b * PLANES * S4 + s4;
    v4f acc = {0.f, 0.f, 0.f, 0.f};
#pragma unroll 4
    for (int c = 0; c < PLANES; ++c) {
        v4f v = __builtin_nontemporal_load(xb + (size_t)c * S4);
        acc += v;
    }
    const float scale = 1.0f / (float)PLANES;
    v4h o;
    o[0] = (_Float16)(acc[0] * scale);
    o[1] = (_Float16)(acc[1] * scale);
    o[2] = (_Float16)(acc[2] * scale);
    o[3] = (_Float16)(acc[3] * scale);
    *(v4h*)(pooledh + (size_t)b * SPATIAL_IN + s4 * 4) = o;
}

// ---------------------------------------------------------------------------
// 2) Weight conversions to f16 with zero padding
// ---------------------------------------------------------------------------
__global__ void conv_w1_kernel(const float* __restrict__ W1,
                               _Float16* __restrict__ W1h) {
    // W1h[o][k], o < 208 (rows >= 196 zero), k < 3136
    const int total = N1PAD * SPATIAL_IN;
    for (int i = blockIdx.x * blockDim.x + threadIdx.x; i < total;
         i += gridDim.x * blockDim.x) {
        int o = i / SPATIAL_IN;
        int k = i - o * SPATIAL_IN;
        float v = (o < HIDDEN) ? W1[o * SPATIAL_IN + k] : 0.0f;
        W1h[i] = (_Float16)v;
    }
}

__global__ void conv_w2_kernel(const float* __restrict__ W2,
                               _Float16* __restrict__ W2h) {
    // W2h[o][k], o < 3136, k < 224 (cols >= 196 zero)
    const int total = SPATIAL_IN * KPAD;
    for (int i = blockIdx.x * blockDim.x + threadIdx.x; i < total;
         i += gridDim.x * blockDim.x) {
        int o = i / KPAD;
        int k = i - o * KPAD;
        float v = (k < HIDDEN) ? W2[o * HIDDEN + k] : 0.0f;
        W2h[i] = (_Float16)v;
    }
}

// ---------------------------------------------------------------------------
// 3) WMMA GEMM: D[m0+16, n0+16] = sum_k A[m,k] * B[n,k]   (B row-major in k)
//    One wave per 16x16 tile. Fragment layouts per CDNA5 ISA 7.12.2:
//      A 16x32 f16: lane<16 -> M=lane, halves 0-7 = K k0+0..7, 8-15 = k0+16..23
//                   lane>=16 -> M=lane-16, K offset +8 / +24
//      B 32x16 f16: lane<16 -> N=lane, halves 0-7 = K k0+0..7, 8-15 = k0+16..23
//                   lane>=16 -> N=lane-16, K offset +8 / +24
//      D f32: lane = n + 16*(m>=8), vgpr = m&7
//    mode 0: relu -> f16 out (ldo stride), mode 1: sigmoid -> f32 out
// ---------------------------------------------------------------------------
__global__ void wmma_gemm_kernel(const _Float16* __restrict__ A, int lda,
                                 const _Float16* __restrict__ Bm, int ldb,
                                 int ktiles,
                                 _Float16* __restrict__ outH,
                                 float* __restrict__ outF,
                                 int ldo, int mode) {
    int lane = threadIdx.x;            // 0..31
    int n0 = blockIdx.x * 16;
    int m0 = blockIdx.y * 16;
    int lrow = lane & 15;
    int hi   = lane >> 4;              // 0 or 1

    const _Float16* Arow = A  + (size_t)(m0 + lrow) * lda + hi * 8;
    const _Float16* Brow = Bm + (size_t)(n0 + lrow) * ldb + hi * 8;

    v8f c = {};
    for (int kt = 0; kt < ktiles; ++kt) {
        int k0 = kt * 32;
        v8h alo = *(const v8h*)(Arow + k0);
        v8h ahi = *(const v8h*)(Arow + k0 + 16);
        v8h blo = *(const v8h*)(Brow + k0);
        v8h bhi = *(const v8h*)(Brow + k0 + 16);
        v16h a, b;
#pragma unroll
        for (int i = 0; i < 8; ++i) {
            a[i] = alo[i];  a[i + 8] = ahi[i];
            b[i] = blo[i];  b[i + 8] = bhi[i];
        }
        c = __builtin_amdgcn_wmma_f32_16x16x32_f16(
                /*neg_a=*/false, a, /*neg_b=*/false, b,
                /*c_mod=*/(short)0, c, /*reuse_a=*/false, /*reuse_b=*/false);
    }

    int n     = lane & 15;
    int mbase = hi * 8;
#pragma unroll
    for (int v = 0; v < 8; ++v) {
        int m = mbase + v;
        float val = c[v];
        size_t idx = (size_t)(m0 + m) * ldo + (n0 + n);
        if (mode == 0) {
            outH[idx] = (_Float16)fmaxf(val, 0.0f);
        } else {
            outF[idx] = 1.0f / (1.0f + __expf(-val));
        }
    }
}

// ---------------------------------------------------------------------------
// 4) Per-row top-k threshold via radix MSB-descent (values all positive =>
//    IEEE bit pattern order == value order). Finds the KEEP-th largest value.
// ---------------------------------------------------------------------------
__global__ void topk_kernel(const float* __restrict__ sig,
                            float* __restrict__ thr) {
    __shared__ unsigned keys[SPATIAL_IN];
    __shared__ int s_cnt;
    int b   = blockIdx.x;
    int tid = threadIdx.x;
    const float* row = sig + (size_t)b * SPATIAL_IN;
    for (int s = tid; s < SPATIAL_IN; s += blockDim.x)
        keys[s] = __float_as_uint(row[s]);
    __syncthreads();

    unsigned prefix = 0u;
    for (int bit = 31; bit >= 0; --bit) {
        unsigned cand = prefix | (1u << bit);
        if (tid == 0) s_cnt = 0;
        __syncthreads();
        int local = 0;
        for (int s = tid; s < SPATIAL_IN; s += blockDim.x)
            local += (keys[s] >= cand) ? 1 : 0;
        atomicAdd(&s_cnt, local);
        __syncthreads();
        if (s_cnt >= KEEP) prefix = cand;   // uniform across block
        __syncthreads();
    }
    if (tid == 0) thr[b] = __uint_as_float(prefix);
}

// ---------------------------------------------------------------------------
// 5) Mask + broadcast to [64,256,3136] f32 (205 MB nontemporal stream write;
//    sig rows stay L2-resident for their 256x reuse).
// ---------------------------------------------------------------------------
__global__ void bcast_kernel(const float* __restrict__ sig,
                             const float* __restrict__ thr,
                             float* __restrict__ out) {
    int t    = blockIdx.x * blockDim.x + threadIdx.x;  // float4 index, = flat/4
    int s4   = t % S4;
    int rest = t / S4;
    int b    = rest / PLANES;
    v4f v  = *(const v4f*)(sig + (size_t)b * SPATIAL_IN + s4 * 4);
    float th = thr[b];
    v4f o;
#pragma unroll
    for (int i = 0; i < 4; ++i) o[i] = (v[i] >= th) ? v[i] : 0.0f;
    __builtin_nontemporal_store(o, (v4f*)out + t);
}

// ---------------------------------------------------------------------------
extern "C" void kernel_launch(void* const* d_in, const int* in_sizes, int n_in,
                              void* d_out, int out_size, void* d_ws, size_t ws_size,
                              hipStream_t stream) {
    const float* x  = (const float*)d_in[0];   // [64,256,56,56]
    const float* W1 = (const float*)d_in[1];   // [196,3136]
    const float* W2 = (const float*)d_in[2];   // [3136,196]
    float* out = (float*)d_out;                // [64,256,56,56]

    char* ws = (char*)d_ws;
    size_t off = 0;
    auto alloc = [&](size_t bytes) {
        char* p = ws + off;
        off += (bytes + 255) & ~(size_t)255;
        return p;
    };
    _Float16* pooledh = (_Float16*)alloc((size_t)BATCH * SPATIAL_IN * 2);
    _Float16* W1h     = (_Float16*)alloc((size_t)N1PAD * SPATIAL_IN * 2);
    _Float16* W2h     = (_Float16*)alloc((size_t)SPATIAL_IN * KPAD * 2);
    _Float16* hh      = (_Float16*)alloc((size_t)BATCH * KPAD * 2);
    float*    sig     = (float*)alloc((size_t)BATCH * SPATIAL_IN * 4);
    float*    thr     = (float*)alloc((size_t)BATCH * 4);

    // zero the K-padded hidden activations (cols 208..223 never written by gemm1)
    hipMemsetAsync(hh, 0, (size_t)BATCH * KPAD * 2, stream);

    // 1) channel-mean pool
    pool_kernel<<<(BATCH * S4) / 256, 256, 0, stream>>>(x, pooledh);

    // 2) weight conversions
    conv_w1_kernel<<<1024, 256, 0, stream>>>(W1, W1h);
    conv_w2_kernel<<<1024, 256, 0, stream>>>(W2, W2h);

    // 3) GEMM1: [64,3136] x W1h[208,3136] -> relu -> hh[64,224] f16
    {
        dim3 grid(N1PAD / 16, BATCH / 16);   // (13, 4)
        wmma_gemm_kernel<<<grid, 32, 0, stream>>>(
            pooledh, SPATIAL_IN, W1h, SPATIAL_IN,
            SPATIAL_IN / 32 /*98*/, hh, nullptr, KPAD, /*mode=*/0);
    }

    // 4) GEMM2: hh[64,224] x W2h[3136,224] -> sigmoid -> sig[64,3136] f32
    {
        dim3 grid(SPATIAL_IN / 16, BATCH / 16);  // (196, 4)
        wmma_gemm_kernel<<<grid, 32, 0, stream>>>(
            hh, KPAD, W2h, KPAD,
            KPAD / 32 /*7*/, nullptr, sig, SPATIAL_IN, /*mode=*/1);
    }

    // 5) per-row top-k threshold
    topk_kernel<<<BATCH, 256, 0, stream>>>(sig, thr);

    // 6) mask + broadcast over channels
    bcast_kernel<<<(BATCH * PLANES * S4) / 256, 256, 0, stream>>>(sig, thr, out);
}
